// PointProposalNet_6390911336590
// MI455X (gfx1250) — matrix operand order repaired and checked by reading, not compiled, not run on hardware
//
#include <hip/hip_runtime.h>
#include <hip/hip_bf16.h>
#include <math.h>
#include <stdint.h>

typedef __attribute__((ext_vector_type(16))) _Float16 v16h;
typedef __attribute__((ext_vector_type(8)))  float    v8f;
typedef _Float16 f16;
typedef __attribute__((address_space(3))) float lds_f32;

#define NPTS   16384   /* NOBJ */
#define KTOP   2048
#define EPSN   1e-5f

/* ---- CDNA5 async global->LDS copy (ASYNCcnt path), inline asm ----------- */
__device__ __forceinline__ void async_ld_f32(float* lds_dst, const float* gsrc)
{
#if defined(__gfx1250__)
    unsigned lofs = (unsigned)(unsigned long long)(lds_f32*)lds_dst;
    asm volatile("global_load_async_to_lds_b32 %0, %1, off"
                 :: "v"(lofs), "v"(gsrc) : "memory");
#else
    *lds_dst = *gsrc;
#endif
}
__device__ __forceinline__ void async_wait0()
{
#if defined(__gfx1250__)
    asm volatile("s_wait_asynccnt 0" ::: "memory");
#endif
}

/* ---------------- selection: stable rank of pts[:,5] (ascending), scatter
   selected rows to obj[rank][0..7] and zero-padded f16 xpad[rank][0..31] --- */
__global__ void __launch_bounds__(256)
k_select(const float* __restrict__ pts, float* __restrict__ obj,
         f16* __restrict__ xpad, int P, int bsel)
{
    __shared__ float sk[256];
    const int tid = threadIdx.x;
    const int i   = blockIdx.x * 256 + tid;
    const size_t base = (size_t)bsel * (size_t)P;
    float ki = (i < P) ? pts[(base + i) * 9 + 5] : 0.f;
    int cnt = 0;
    for (int t0 = 0; t0 < P; t0 += 256) {
        int j = t0 + tid;
        if (j < P) async_ld_f32(&sk[tid], pts + (base + j) * 9 + 5);
        async_wait0();
        __syncthreads();
        if (i < P) {
            int lim = (P - t0) < 256 ? (P - t0) : 256;
            for (int jj = 0; jj < lim; ++jj) {
                float kj = sk[jj];
                int   jg = t0 + jj;
                cnt += (kj < ki) || (kj == ki && jg < i);
            }
        }
        __syncthreads();
    }
    if (i < P && cnt < NPTS) {
#pragma unroll
        for (int c = 0; c < 8; ++c) {
            float v = pts[(base + i) * 9 + 1 + c];
            obj[(size_t)cnt * 8 + c]   = v;
            xpad[(size_t)cnt * 32 + c] = (f16)v;
        }
#pragma unroll
        for (int c = 8; c < 32; ++c) xpad[(size_t)cnt * 32 + c] = (f16)0.f;
    }
}

/* ---------------- f32 weight [Nout][Kin] -> f16 [Nout][Kp] (zero padded) --- */
__global__ void __launch_bounds__(256)
k_convert_w(const float* __restrict__ w, f16* __restrict__ w16,
            int Nout, int Kin, int Kp)
{
    size_t idx = (size_t)blockIdx.x * 256 + threadIdx.x;
    size_t total = (size_t)Nout * Kp;
    if (idx >= total) return;
    int k = (int)(idx % Kp);
    size_t n = idx / Kp;
    w16[idx] = (k < Kin) ? (f16)w[n * Kin + k] : (f16)0.f;
}

/* ---------------- register-blocked WMMA GEMM:
   out[M][Nout] = A16[M][Kp] * W16[Nout][Kp]^T + bias
   each wave computes a 32x64 block (2x4 tiles of 16x16), 8 WMMAs / k-step -- */
__global__ void __launch_bounds__(256)
k_gemm_wmma(const f16* __restrict__ A, const f16* __restrict__ W,
            const float* __restrict__ bias, float* __restrict__ out,
            int M, int Kp, int Nout)
{
    const int lane = threadIdx.x & 31;
    const int wave = (blockIdx.x * blockDim.x + threadIdx.x) >> 5;
    const int ntb = Nout >> 6;              /* 64-wide n blocks  */
    const int tiles = (M >> 5) * ntb;       /* 32-tall m blocks  */
    if (wave >= tiles) return;              /* wave-uniform      */
    const int bm = wave / ntb, bn = wave % ntb;
    const int lm = lane & 15, lh = lane >> 4;
    const f16* arow0 = A + (size_t)(bm * 32 + lm) * Kp;
    const f16* arow1 = arow0 + (size_t)16 * Kp;
    const f16* wbase = W + (size_t)(bn * 64 + lm) * Kp;
    v8f acc[2][4] = {};
    for (int k0 = 0; k0 < Kp; k0 += 32) {
        __builtin_prefetch(arow0 + k0 + 128, 0, 0);
        __builtin_prefetch(arow1 + k0 + 128, 0, 0);
        v16h a0, a1;
        {
            const f16* ap0 = arow0 + k0 + lh * 16;
            const f16* ap1 = arow1 + k0 + lh * 16;
#pragma unroll
            for (int j = 0; j < 8; ++j) {
                a0[j] = ap0[j]; a0[j + 8] = ap0[j + 8];
                a1[j] = ap1[j]; a1[j + 8] = ap1[j + 8];
            }
        }
#pragma unroll
        for (int jt = 0; jt < 4; ++jt) {
            v16h b;
            const f16* bp = wbase + (size_t)jt * 16 * Kp + k0 + lh * 16;
#pragma unroll
            for (int j = 0; j < 16; ++j) b[j] = bp[j];
            acc[0][jt] = __builtin_amdgcn_wmma_f32_16x16x32_f16(
                false, a0, false, b, (short)0, acc[0][jt], false, false);
            acc[1][jt] = __builtin_amdgcn_wmma_f32_16x16x32_f16(
                false, a1, false, b, (short)0, acc[1][jt], false, false);
        }
    }
#pragma unroll
    for (int mi = 0; mi < 2; ++mi) {
#pragma unroll
        for (int jt = 0; jt < 4; ++jt) {
#pragma unroll
            for (int v = 0; v < 8; ++v) {
                int r = bm * 32 + mi * 16 + v + lh * 8;
                int c = bn * 64 + jt * 16 + lm;
                out[(size_t)r * Nout + c] = acc[mi][jt][v] + bias[c];
            }
        }
    }
}

/* ---------------- fused gate/filter dual-WMMA GEMM (2x2 blocking):
   r16 = sigmoid(A*Wg^T + wg_b + b_gate) * (A*Wf^T + wf_b) ------------------ */
__global__ void __launch_bounds__(256)
k_gemm_gate(const f16* __restrict__ A, const f16* __restrict__ Wg,
            const f16* __restrict__ Wf, const float* __restrict__ bg,
            const float* __restrict__ bgate, const float* __restrict__ bf,
            f16* __restrict__ out, int M, int Kp, int Nout)
{
    const int lane = threadIdx.x & 31;
    const int wave = (blockIdx.x * blockDim.x + threadIdx.x) >> 5;
    const int ntb = Nout >> 5;              /* 32-wide n blocks */
    const int tiles = (M >> 5) * ntb;       /* 32-tall m blocks */
    if (wave >= tiles) return;
    const int bm = wave / ntb, bn = wave % ntb;
    const int lm = lane & 15, lh = lane >> 4;
    const f16* arow0 = A + (size_t)(bm * 32 + lm) * Kp;
    const f16* arow1 = arow0 + (size_t)16 * Kp;
    const f16* gbase = Wg + (size_t)(bn * 32 + lm) * Kp;
    const f16* fbase = Wf + (size_t)(bn * 32 + lm) * Kp;
    v8f accg[2][2] = {}, accf[2][2] = {};
    for (int k0 = 0; k0 < Kp; k0 += 32) {
        __builtin_prefetch(arow0 + k0 + 128, 0, 0);
        v16h a0, a1;
        {
            const f16* ap0 = arow0 + k0 + lh * 16;
            const f16* ap1 = arow1 + k0 + lh * 16;
#pragma unroll
            for (int j = 0; j < 8; ++j) {
                a0[j] = ap0[j]; a0[j + 8] = ap0[j + 8];
                a1[j] = ap1[j]; a1[j + 8] = ap1[j + 8];
            }
        }
#pragma unroll
        for (int jt = 0; jt < 2; ++jt) {
            v16h b;
            const f16* bp = gbase + (size_t)jt * 16 * Kp + k0 + lh * 16;
#pragma unroll
            for (int j = 0; j < 16; ++j) b[j] = bp[j];
            accg[0][jt] = __builtin_amdgcn_wmma_f32_16x16x32_f16(
                false, a0, false, b, (short)0, accg[0][jt], false, false);
            accg[1][jt] = __builtin_amdgcn_wmma_f32_16x16x32_f16(
                false, a1, false, b, (short)0, accg[1][jt], false, false);
        }
#pragma unroll
        for (int jt = 0; jt < 2; ++jt) {
            v16h b;
            const f16* bp = fbase + (size_t)jt * 16 * Kp + k0 + lh * 16;
#pragma unroll
            for (int j = 0; j < 16; ++j) b[j] = bp[j];
            accf[0][jt] = __builtin_amdgcn_wmma_f32_16x16x32_f16(
                false, a0, false, b, (short)0, accf[0][jt], false, false);
            accf[1][jt] = __builtin_amdgcn_wmma_f32_16x16x32_f16(
                false, a1, false, b, (short)0, accf[1][jt], false, false);
        }
    }
#pragma unroll
    for (int mi = 0; mi < 2; ++mi) {
#pragma unroll
        for (int jt = 0; jt < 2; ++jt) {
#pragma unroll
            for (int v = 0; v < 8; ++v) {
                int r = bm * 32 + mi * 16 + v + lh * 8;
                int c = bn * 32 + jt * 16 + lm;
                float g  = accg[mi][jt][v] + bg[c] + bgate[c];
                float fv = accf[mi][jt][v] + bf[c];
                float sg = 1.f / (1.f + __expf(-g));
                out[(size_t)r * Nout + c] = (f16)(sg * fv);
            }
        }
    }
}

/* ---------------- per-channel (column) mean / rstd over M rows ------------ */
__global__ void __launch_bounds__(256)
k_colstats(const float* __restrict__ x, float* __restrict__ mean,
           float* __restrict__ rstd, int M, int C)
{
    __shared__ float rs[256], rs2[256];
    const int c = blockIdx.x, tid = threadIdx.x;
    float s = 0.f, s2 = 0.f;
    for (int n = tid; n < M; n += 256) {
        float v = x[(size_t)n * C + c];
        s += v; s2 += v * v;
    }
    rs[tid] = s; rs2[tid] = s2; __syncthreads();
    for (int st = 128; st > 0; st >>= 1) {
        if (tid < st) { rs[tid] += rs[tid + st]; rs2[tid] += rs2[tid + st]; }
        __syncthreads();
    }
    if (tid == 0) {
        float m = rs[0] / (float)M;
        float v = rs2[0] / (float)M - m * m;
        mean[c] = m;
        rstd[c] = rsqrtf(v + EPSN);
    }
}

/* ---------------- per-channel normalize + relu -> f16 --------------------- */
__global__ void __launch_bounds__(256)
k_colnorm(const float* __restrict__ x, const float* __restrict__ mean,
          const float* __restrict__ rstd, f16* __restrict__ out,
          int M, int C)
{
    size_t idx = (size_t)blockIdx.x * 256 + threadIdx.x;
    if (idx >= (size_t)M * C) return;
    int c = (int)(idx % C);
    float v = (x[idx] - mean[c]) * rstd[c];
    out[idx] = (f16)(v > 0.f ? v : 0.f);
}

/* ---------------- per-channel max of normalized value (opt relu) ---------- */
__global__ void __launch_bounds__(256)
k_colmax(const float* __restrict__ x, const float* __restrict__ mean,
         const float* __restrict__ rstd, float* __restrict__ g,
         int M, int C, int do_relu)
{
    __shared__ float rm[256];
    const int c = blockIdx.x, tid = threadIdx.x;
    const float m = mean[c], r = rstd[c];
    float best = -3.4e38f;
    for (int n = tid; n < M; n += 256) {
        float v = (x[(size_t)n * C + c] - m) * r;
        best = v > best ? v : best;
    }
    rm[tid] = best; __syncthreads();
    for (int st = 128; st > 0; st >>= 1) {
        if (tid < st) rm[tid] = rm[tid] > rm[tid + st] ? rm[tid] : rm[tid + st];
        __syncthreads();
    }
    if (tid == 0) {
        float v = rm[0];
        if (do_relu && v < 0.f) v = 0.f;
        g[c] = v;
    }
}

/* ---------------- small FC on a vector + instance-norm + relu ------------- */
__global__ void __launch_bounds__(256)
k_fc_norm_relu(const float* __restrict__ in, const float* __restrict__ w,
               const float* __restrict__ b, float* __restrict__ out,
               int Kin, int Cout)
{
    __shared__ float buf[512];
    __shared__ float rs[256], rs2[256];
    const int tid = threadIdx.x;
    for (int c = tid; c < Cout; c += 256) {
        float y = b[c];
        for (int k = 0; k < Kin; ++k) y += w[(size_t)c * Kin + k] * in[k];
        buf[c] = y;
    }
    __syncthreads();
    float s = 0.f, s2 = 0.f;
    for (int c = tid; c < Cout; c += 256) { float y = buf[c]; s += y; s2 += y * y; }
    rs[tid] = s; rs2[tid] = s2; __syncthreads();
    for (int st = 128; st > 0; st >>= 1) {
        if (tid < st) { rs[tid] += rs[tid + st]; rs2[tid] += rs2[tid + st]; }
        __syncthreads();
    }
    float m  = rs[0] / (float)Cout;
    float vv = rs2[0] / (float)Cout - m * m;
    float r  = rsqrtf(vv + EPSN);
    __syncthreads();
    for (int c = tid; c < Cout; c += 256) {
        float y = (buf[c] - m) * r;
        out[c] = y > 0.f ? y : 0.f;
    }
}

/* ---------------- final STN FC: trans = f3w@g + f3b + eye(8) -------------- */
__global__ void k_fc_trans(const float* __restrict__ in, const float* __restrict__ w,
                           const float* __restrict__ b, float* __restrict__ trans)
{
    const int c = threadIdx.x;
    if (c < 64) {
        float y = b[c];
        for (int k = 0; k < 256; ++k) y += w[c * 256 + k] * in[k];
        trans[c] = y + (((c >> 3) == (c & 7)) ? 1.f : 0.f);
    }
}

/* ---------------- x' = trans^T @ x, point-major, f16 K-pad to 32 ---------- */
__global__ void __launch_bounds__(256)
k_apply_trans(const float* __restrict__ obj, const float* __restrict__ trans,
              f16* __restrict__ xpad, int N)
{
    size_t idx = (size_t)blockIdx.x * 256 + threadIdx.x;
    if (idx >= (size_t)N * 32) return;
    int c = (int)(idx & 31);
    size_t n = idx >> 5;
    float v = 0.f;
    if (c < 8) {
#pragma unroll
        for (int ic = 0; ic < 8; ++ic) v += obj[n * 8 + ic] * trans[ic * 8 + c];
    }
    xpad[idx] = (f16)v;
}

/* ---------------- feat16[n] = [gmax(1024) ; pf16[n](64)] ------------------ */
__global__ void __launch_bounds__(256)
k_featbuild(const float* __restrict__ gmax, const f16* __restrict__ pf,
            f16* __restrict__ feat, int N)
{
    size_t idx = (size_t)blockIdx.x * 256 + threadIdx.x;
    if (idx >= (size_t)N * 1088) return;
    int c = (int)(idx % 1088);
    size_t n = idx / 1088;
    feat[idx] = (c < 1024) ? (f16)gmax[c] : pf[n * 64 + (c - 1024)];
}

/* ---------------- row-wise instance-norm + relu -> f16 (block per row) ---- */
__global__ void __launch_bounds__(256)
k_rownorm_relu(const float* __restrict__ x, f16* __restrict__ out, int C)
{
    __shared__ float rs[256], rs2[256];
    const int n = blockIdx.x, tid = threadIdx.x;
    const float* row = x + (size_t)n * C;
    float s = 0.f, s2 = 0.f;
    for (int c = tid; c < C; c += 256) { float v = row[c]; s += v; s2 += v * v; }
    rs[tid] = s; rs2[tid] = s2; __syncthreads();
    for (int st = 128; st > 0; st >>= 1) {
        if (tid < st) { rs[tid] += rs[tid + st]; rs2[tid] += rs2[tid + st]; }
        __syncthreads();
    }
    float m  = rs[0] / (float)C;
    float vv = rs2[0] / (float)C - m * m;
    float r  = rsqrtf(vv + EPSN);
    __syncthreads();
    for (int c = tid; c < C; c += 256) {
        float v = (row[c] - m) * r;
        out[(size_t)n * C + c] = (f16)(v > 0.f ? v : 0.f);
    }
}

/* ---------------- fused e2-rownorm-relu + e3 dot -> sf[n] (C == 256) ------ */
__global__ void __launch_bounds__(256)
k_row_e2e3(const float* __restrict__ x, const float* __restrict__ e3w,
           const float* __restrict__ e3b, float* __restrict__ sf)
{
    __shared__ float rs[256], rs2[256];
    const int n = blockIdx.x, tid = threadIdx.x;
    const float v = x[(size_t)n * 256 + tid];
    rs[tid] = v; rs2[tid] = v * v; __syncthreads();
    for (int st = 128; st > 0; st >>= 1) {
        if (tid < st) { rs[tid] += rs[tid + st]; rs2[tid] += rs2[tid + st]; }
        __syncthreads();
    }
    float m  = rs[0] / 256.f;
    float vv = rs2[0] / 256.f - m * m;
    float r  = rsqrtf(vv + EPSN);
    float u  = (v - m) * r;
    u = u > 0.f ? u : 0.f;
    __syncthreads();
    rs[tid] = u * e3w[tid]; __syncthreads();
    for (int st = 128; st > 0; st >>= 1) {
        if (tid < st) rs[tid] += rs[tid + st];
        __syncthreads();
    }
    if (tid == 0) sf[n] = rs[0] + e3b[0];
}

/* ---------------- stable descending rank; scatter top-K to d_out ---------- */
__global__ void __launch_bounds__(256)
k_rank_out(const float* __restrict__ sf, const float* __restrict__ obj,
           float* __restrict__ outp, int N, int K)
{
    __shared__ float sk[256];
    const int tid = threadIdx.x;
    const int i   = blockIdx.x * 256 + tid;
    const float ki = (i < N) ? sf[i] : 0.f;
    int cnt = 0;
    for (int t0 = 0; t0 < N; t0 += 256) {
        int j = t0 + tid;
        if (j < N) async_ld_f32(&sk[tid], sf + j);
        async_wait0();
        __syncthreads();
        if (i < N) {
            int lim = (N - t0) < 256 ? (N - t0) : 256;
            for (int jj = 0; jj < lim; ++jj) {
                float kj = sk[jj];
                int   jg = t0 + jj;
                cnt += (kj > ki) || (kj == ki && jg < i);
            }
        }
        __syncthreads();
    }
    if (i < N && cnt < K) {
        outp[cnt * 3 + 0] = obj[(size_t)i * 8 + 0];
        outp[cnt * 3 + 1] = obj[(size_t)i * 8 + 1];
        outp[cnt * 3 + 2] = obj[(size_t)i * 8 + 2];
        outp[(size_t)K * 3 + cnt] = ki;
    }
}

/* ========================================================================= */
extern "C" void kernel_launch(void* const* d_in, const int* in_sizes, int n_in,
                              void* d_out, int out_size, void* d_ws, size_t ws_size,
                              hipStream_t stream)
{
    (void)n_in; (void)out_size; (void)ws_size;
    const float* pts      = (const float*)d_in[0];
    const float* stn_c1w  = (const float*)d_in[1];
    const float* stn_c1b  = (const float*)d_in[2];
    const float* stn_c2w  = (const float*)d_in[3];
    const float* stn_c2b  = (const float*)d_in[4];
    const float* stn_c3w  = (const float*)d_in[5];
    const float* stn_c3b  = (const float*)d_in[6];
    const float* stn_f1w  = (const float*)d_in[7];
    const float* stn_f1b  = (const float*)d_in[8];
    const float* stn_f2w  = (const float*)d_in[9];
    const float* stn_f2b  = (const float*)d_in[10];
    const float* stn_f3w  = (const float*)d_in[11];
    const float* stn_f3b  = (const float*)d_in[12];
    const float* c1w      = (const float*)d_in[13];
    const float* c1b      = (const float*)d_in[14];
    const float* c2w      = (const float*)d_in[15];
    const float* c2b      = (const float*)d_in[16];
    const float* c3w      = (const float*)d_in[17];
    const float* c3b      = (const float*)d_in[18];
    const float* wg_w     = (const float*)d_in[19];
    const float* wg_b     = (const float*)d_in[20];
    const float* b_gate   = (const float*)d_in[21];
    const float* wf_w     = (const float*)d_in[22];
    const float* wf_b     = (const float*)d_in[23];
    const float* e1w      = (const float*)d_in[24];
    const float* e1b      = (const float*)d_in[25];
    const float* e2w      = (const float*)d_in[26];
    const float* e2b      = (const float*)d_in[27];
    const float* e3w      = (const float*)d_in[28];
    const float* e3b      = (const float*)d_in[29];

    const int Bsz  = 2;                         /* module constant B */
    const int P    = (in_sizes[0] / 9) / Bsz;   /* points per batch  */
    const int bsel = Bsz - 1;                   /* only last batch returned */
    const int M    = NPTS;

    /* ---- workspace carve-out (256B aligned) ---- */
    char*  base = (char*)d_ws;
    size_t off  = 0;
    auto take = [&](size_t bytes) -> void* {
        void* p = base + off;
        off = (off + bytes + 255) & ~(size_t)255;
        return p;
    };
    float* obj    = (float*)take((size_t)M * 8 * 4);
    f16*   actA   = (f16*)  take((size_t)M * 1088 * 2);
    f16*   actB   = (f16*)  take((size_t)M * 1088 * 2);
    float* f32a   = (float*)take((size_t)M * 1088 * 4);
    f16*   w16a   = (f16*)  take((size_t)1088 * 1088 * 2);
    f16*   w16b   = (f16*)  take((size_t)1088 * 1088 * 2);
    float* meanp  = (float*)take(1088 * 4);
    float* rstdp  = (float*)take(1088 * 4);
    float* gvec   = (float*)take(1024 * 4);
    float* fc1    = (float*)take(512 * 4);
    float* fc2    = (float*)take(256 * 4);
    float* transp = (float*)take(64 * 4);
    float* gmaxv  = (float*)take(1024 * 4);
    float* sfp    = (float*)take((size_t)M * 4);

    auto gemm = [&](const f16* A, const float* Wsrc, const float* bias,
                    int Kin, int Kp, int Nout) {
        size_t wtot = (size_t)Nout * Kp;
        k_convert_w<<<(unsigned)((wtot + 255) / 256), 256, 0, stream>>>(
            Wsrc, w16a, Nout, Kin, Kp);
        int tiles = (M / 32) * (Nout / 64);
        k_gemm_wmma<<<(unsigned)((tiles + 7) / 8), 256, 0, stream>>>(
            A, w16a, bias, f32a, M, Kp, Nout);
    };
    auto norm_relu = [&](int C, f16* dst) {
        k_colstats<<<(unsigned)C, 256, 0, stream>>>(f32a, meanp, rstdp, M, C);
        size_t tot = (size_t)M * C;
        k_colnorm<<<(unsigned)((tot + 255) / 256), 256, 0, stream>>>(
            f32a, meanp, rstdp, dst, M, C);
    };

    /* ---- select batch `bsel`, stable sort by feature 4, take first NOBJ -- */
    k_select<<<(unsigned)((P + 255) / 256), 256, 0, stream>>>(pts, obj, actA, P, bsel);

    /* ---- STN conv stack ---- */
    gemm(actA, stn_c1w, stn_c1b,   8,  32,   64); norm_relu(64,  actB);
    gemm(actB, stn_c2w, stn_c2b,  64,  64,  128); norm_relu(128, actA);
    gemm(actA, stn_c3w, stn_c3b, 128, 128, 1024);
    k_colstats<<<1024, 256, 0, stream>>>(f32a, meanp, rstdp, M, 1024);
    k_colmax<<<1024, 256, 0, stream>>>(f32a, meanp, rstdp, gvec, M, 1024, 1);

    /* ---- STN FC head -> 8x8 transform ---- */
    k_fc_norm_relu<<<1, 256, 0, stream>>>(gvec, stn_f1w, stn_f1b, fc1, 1024, 512);
    k_fc_norm_relu<<<1, 256, 0, stream>>>(fc1,  stn_f2w, stn_f2b, fc2,  512, 256);
    k_fc_trans<<<1, 64, 0, stream>>>(fc2, stn_f3w, stn_f3b, transp);
    k_apply_trans<<<(unsigned)(((size_t)M * 32 + 255) / 256), 256, 0, stream>>>(
        obj, transp, actA, M);

    /* ---- main conv stack ---- */
    gemm(actA, c1w, c1b,   8,  32,   64); norm_relu(64,  actB);  /* actB = pf16 */
    gemm(actB, c2w, c2b,  64,  64,  128); norm_relu(128, actA);
    gemm(actA, c3w, c3b, 128, 128, 1024);
    k_colstats<<<1024, 256, 0, stream>>>(f32a, meanp, rstdp, M, 1024);
    k_colmax<<<1024, 256, 0, stream>>>(f32a, meanp, rstdp, gmaxv, M, 1024, 0);

    /* ---- feat = [gmax ; pf] (f16), fused gate/filter dual-WMMA GEMM ---- */
    k_featbuild<<<(unsigned)(((size_t)M * 1088 + 255) / 256), 256, 0, stream>>>(
        gmaxv, actB, actA, M);
    {
        size_t wtot = (size_t)1088 * 1088;
        k_convert_w<<<(unsigned)((wtot + 255) / 256), 256, 0, stream>>>(
            wg_w, w16a, 1088, 1088, 1088);
        k_convert_w<<<(unsigned)((wtot + 255) / 256), 256, 0, stream>>>(
            wf_w, w16b, 1088, 1088, 1088);
        int tiles = (M / 32) * (1088 / 32);
        k_gemm_gate<<<(unsigned)((tiles + 7) / 8), 256, 0, stream>>>(
            actA, w16a, w16b, wg_b, b_gate, wf_b, actB /* r16 */, M, 1088, 1088);
    }

    /* ---- scoring head ---- */
    gemm(actB, e1w, e1b, 1088, 1088, 512);
    k_rownorm_relu<<<(unsigned)M, 256, 0, stream>>>(f32a, actA, 512);
    gemm(actA, e2w, e2b, 512, 512, 256);
    k_row_e2e3<<<(unsigned)M, 256, 0, stream>>>(f32a, e3w, e3b, sfp);

    /* ---- top-K by descending score -> d_out [K*3 coords | K scores] ---- */
    k_rank_out<<<(unsigned)(M / 256), 256, 0, stream>>>(
        sfp, obj, (float*)d_out, M, KTOP);
}
